// FastFeedForwardNetwork_48773648613626
// MI455X (gfx1250) — matrix operand order repaired and checked by reading, not compiled, not run on hardware
//
#include <hip/hip_runtime.h>
#include <hip/hip_bf16.h>
#include <math.h>

// ---------------------------------------------------------------------------
// FastFeedForwardNetwork for MI455X (gfx1250, wave32, WMMA)
// B=4, T=2048 -> 8192 tokens, D=1024, 16 leaves x LEAF_DIM=256.
//   probs  : (8192,16) f32
//   GEMM1  : (8192x1024) @ (1024x4096) -> H (relu+prob scaled, f16)
//   GEMM2  : (8192x4096) @ (4096x1024) -> out (+ probs@b2, f32)
// Register-staged pipelined global->LDS so global loads overlap the WMMAs.
// ---------------------------------------------------------------------------

typedef __attribute__((ext_vector_type(16))) _Float16 v16h;
typedef __attribute__((ext_vector_type(8)))  _Float16 v8h;
typedef __attribute__((ext_vector_type(4)))  _Float16 v4h;
typedef __attribute__((ext_vector_type(8)))  float    v8f;

union V16U { v16h v; v8h h[2]; };

static constexpr int TOK  = 8192;   // B*T
static constexpr int DIM  = 1024;   // D
static constexpr int LEAF = 256;    // LEAF_DIM
static constexpr int NL   = 16;     // leaves
static constexpr int NH   = NL * LEAF;   // 4096 hidden cols (concat)

// ---------------- x -> f16 ----------------
__global__ void ffn_convert_x(const float4* __restrict__ in,
                              v4h* __restrict__ out, int n4) {
  int i = blockIdx.x * blockDim.x + threadIdx.x;
  if (i < n4) {
    float4 f = in[i];
    v4h h = { (_Float16)f.x, (_Float16)f.y, (_Float16)f.z, (_Float16)f.w };
    out[i] = h;
  }
}

// ---------------- weight transpose + f16 convert ----------------
// in:  per-leaf row-major (K rows x N cols) f32 at in + l*K*N
// out: element (l, k, n) written to out[(l*rowBaseL + n)*ldo + l*colBaseL + k]
__global__ void ffn_transpose_cvt(const float* __restrict__ in,
                                  _Float16* __restrict__ out,
                                  int K, int N, int ldo,
                                  int rowBaseL, int colBaseL) {
  __shared__ float tile[32][33];
  const int l  = blockIdx.z;
  const int k0 = blockIdx.y * 32;
  const int n0 = blockIdx.x * 32;
  const float* inl = in + (size_t)l * K * N;
  for (int r = threadIdx.y; r < 32; r += 8)
    tile[r][threadIdx.x] = inl[(size_t)(k0 + r) * N + n0 + threadIdx.x];
  __syncthreads();
  const int rowB = l * rowBaseL + n0;
  const int colB = l * colBaseL + k0;
  for (int r = threadIdx.y; r < 32; r += 8)
    out[(size_t)(rowB + r) * ldo + colB + threadIdx.x] =
        (_Float16)tile[threadIdx.x][r];
}

// ---------------- leaf probabilities ----------------
// One wave (32 lanes) per token: 15 length-1024 dots, sigmoid, tree product.
__global__ __launch_bounds__(256) void ffn_probs(
    const float* __restrict__ x, const float* __restrict__ node_w,
    const float* __restrict__ node_b, float* __restrict__ probs) {
  __shared__ float cs[8][16];
  const int lane = threadIdx.x & 31;
  const int wid  = threadIdx.x >> 5;
  const int t    = blockIdx.x * 8 + wid;   // grid sized exactly -> always valid

  float xv[32];
  const float* xr = x + (size_t)t * DIM;
#pragma unroll
  for (int j = 0; j < 32; ++j) xv[j] = xr[lane + 32 * j];

#pragma unroll
  for (int n = 0; n < 15; ++n) {
    const float* wr = node_w + n * DIM;
    float s = 0.f;
#pragma unroll
    for (int j = 0; j < 32; ++j) s += xv[j] * wr[lane + 32 * j];
#pragma unroll
    for (int off = 16; off > 0; off >>= 1) s += __shfl_xor(s, off, 32);
    if (lane == 0) cs[wid][n] = 1.0f / (1.0f + __expf(-(s + node_b[n])));
  }
  __syncthreads();

  if (lane < 16) {
    const int leaf = lane;
    float p = 1.f;
#pragma unroll
    for (int m = 0; m < 4; ++m) {
      int prefix = leaf >> (4 - m);
      int node   = (1 << m) - 1 + prefix;
      int bit    = (leaf >> (3 - m)) & 1;
      float c    = cs[wid][node];
      p *= bit ? (1.0f - c) : c;
    }
    probs[t * NL + leaf] = p;
  }
}

// ---------------- WMMA GEMM, C = A(MxK) * Bt(NxK)^T ----------------
// MODE 0: Hout = relu(C + b1[col]) * probs[row][col>>8]            (f16)
// MODE 1: Fout = C + sum_l probs[row][l] * b2[l*N + col]           (f32)
template <int MODE>
__global__ __launch_bounds__(256) void ffn_gemm(
    const _Float16* __restrict__ A, const _Float16* __restrict__ Bt,
    const float* __restrict__ probs, const float* __restrict__ bias,
    _Float16* __restrict__ Hout, float* __restrict__ Fout,
    int M, int N, int K) {
  constexpr int BM = 128, BN = 128, BK = 64, LDL = BK + 8;  // +8 halves pad
  __shared__ alignas(16) _Float16 As[BM][LDL];
  __shared__ alignas(16) _Float16 Bs[BN][LDL];

  const int tid  = threadIdx.x;
  const int lane = tid & 31;
  const int wave = tid >> 5;
  const int wm   = wave & 3;   // 4 waves along M (32 rows each)
  const int wn   = wave >> 2;  // 2 waves along N (64 cols each)
  const int Mb   = blockIdx.y * BM;
  const int Nb   = blockIdx.x * BN;

  const v8f vzero = {0.f, 0.f, 0.f, 0.f, 0.f, 0.f, 0.f, 0.f};
  v8f acc[2][4];
#pragma unroll
  for (int j = 0; j < 2; ++j)
#pragma unroll
    for (int i = 0; i < 4; ++i) acc[j][i] = vzero;

  // Per-thread staging chunk coordinates (4 chunks of 8 halves per matrix).
  const int rr0 = tid >> 3;            // row for chunk it: rr0 + 32*it
  const int cc0 = (tid & 7) << 3;      // col offset within 64-half row
  const _Float16* Abase  = A  + (size_t)(Mb + rr0) * K + cc0;
  const _Float16* Btbase = Bt + (size_t)(Nb + rr0) * K + cc0;

  v8h rA[4], rB[4];  // register-staged tile (keeps global loads in flight)

  // ---- prologue: tile 0 ----
#pragma unroll
  for (int it = 0; it < 4; ++it) {
    rA[it] = *(const v8h*)(Abase  + (size_t)(32 * it) * K);
    rB[it] = *(const v8h*)(Btbase + (size_t)(32 * it) * K);
  }
#pragma unroll
  for (int it = 0; it < 4; ++it) {
    *(v8h*)&As[rr0 + 32 * it][cc0] = rA[it];
    *(v8h*)&Bs[rr0 + 32 * it][cc0] = rB[it];
  }
  __syncthreads();

  for (int k0 = 0; k0 < K; k0 += BK) {
    const bool has_next = (k0 + BK) < K;
    // Issue next tile's global loads up front; they complete while we WMMA.
    if (has_next) {
#pragma unroll
      for (int it = 0; it < 4; ++it) {
        rA[it] = *(const v8h*)(Abase  + (size_t)(32 * it) * K + (k0 + BK));
        rB[it] = *(const v8h*)(Btbase + (size_t)(32 * it) * K + (k0 + BK));
      }
      if (k0 + 2 * BK < K) {  // warm L2 one more tile ahead
        __builtin_prefetch(Abase  + (size_t)rr0 * 0 + (k0 + 2 * BK), 0, 1);
        __builtin_prefetch(Btbase + (k0 + 2 * BK), 0, 1);
      }
    }

    // ---- compute current LDS tile ----
#pragma unroll
    for (int ks = 0; ks < BK; ks += 32) {
      V16U a[2], b[4];
      // A frag (16x32): lane<16 -> row=lane, K 0-7 & 16-23; lane>=16 -> K 8-15 & 24-31
      const int ra = wm * 32 + (lane & 15);
      const int ka = ks + ((lane >> 4) << 3);
#pragma unroll
      for (int j = 0; j < 2; ++j) {
        a[j].h[0] = *(const v8h*)&As[ra + 16 * j][ka];
        a[j].h[1] = *(const v8h*)&As[ra + 16 * j][ka + 16];
      }
      // B frag (32x16): col = lane&15; lane<16 -> K 0-7 & 16-23; lane>=16 -> K 8-15 & 24-31
      const int cb = wn * 64 + (lane & 15);
      const int kb = ks + ((lane >> 4) << 3);
#pragma unroll
      for (int i = 0; i < 4; ++i) {
        b[i].h[0] = *(const v8h*)&Bs[cb + 16 * i][kb];
        b[i].h[1] = *(const v8h*)&Bs[cb + 16 * i][kb + 16];
      }
#pragma unroll
      for (int j = 0; j < 2; ++j)
#pragma unroll
        for (int i = 0; i < 4; ++i)
          acc[j][i] = __builtin_amdgcn_wmma_f32_16x16x32_f16(
              false, a[j].v, false, b[i].v, (short)0, acc[j][i], false, false);
    }
    __syncthreads();  // everyone done reading the LDS tile

    if (has_next) {   // commit staged registers (single wait here, not per-load)
#pragma unroll
      for (int it = 0; it < 4; ++it) {
        *(v8h*)&As[rr0 + 32 * it][cc0] = rA[it];
        *(v8h*)&Bs[rr0 + 32 * it][cc0] = rB[it];
      }
      __syncthreads();
    }
  }

  // Epilogue.  C/D layout: vgpr v, lane -> col = lane&15, row = v + 8*(lane>>4)
  const int colL = lane & 15;
  const int rsel = (lane >> 4) * 8;
#pragma unroll
  for (int j = 0; j < 2; ++j) {
#pragma unroll
    for (int i = 0; i < 4; ++i) {
      const int col = Nb + wn * 64 + i * 16 + colL;
      if (MODE == 0) {
        const float bv   = bias[col];       // b1 flat (16*256)
        const int   leaf = col >> 8;
#pragma unroll
        for (int v = 0; v < 8; ++v) {
          const int row = Mb + wm * 32 + j * 16 + rsel + v;
          float val = acc[j][i][v] + bv;
          val = fmaxf(val, 0.f) * probs[row * NL + leaf];
          Hout[(size_t)row * N + col] = (_Float16)val;
        }
      } else {
        float b2c[16];
#pragma unroll
        for (int l = 0; l < 16; ++l) b2c[l] = bias[l * N + col];  // b2 (16,1024)
#pragma unroll
        for (int v = 0; v < 8; ++v) {
          const int row = Mb + wm * 32 + j * 16 + rsel + v;
          float s = acc[j][i][v];
          const float4* pr4 = (const float4*)(probs + row * NL);
#pragma unroll
          for (int q = 0; q < 4; ++q) {
            float4 p = pr4[q];
            s += p.x * b2c[4 * q + 0];
            s += p.y * b2c[4 * q + 1];
            s += p.z * b2c[4 * q + 2];
            s += p.w * b2c[4 * q + 3];
          }
          Fout[(size_t)row * N + col] = s;
        }
      }
    }
  }
}

// ---------------------------------------------------------------------------
extern "C" void kernel_launch(void* const* d_in, const int* in_sizes, int n_in,
                              void* d_out, int out_size, void* d_ws,
                              size_t ws_size, hipStream_t stream) {
  (void)in_sizes; (void)n_in; (void)out_size; (void)ws_size;
  const float* x      = (const float*)d_in[0];
  const float* node_w = (const float*)d_in[1];
  const float* node_b = (const float*)d_in[2];
  const float* w1     = (const float*)d_in[3];
  const float* b1     = (const float*)d_in[4];
  const float* w2     = (const float*)d_in[5];
  const float* b2     = (const float*)d_in[6];
  float* out = (float*)d_out;

  // workspace carve-up (~97 MB total)
  char* ws = (char*)d_ws;
  _Float16* xh  = (_Float16*)ws; ws += (size_t)TOK * DIM * 2;        // 16 MB
  _Float16* w1t = (_Float16*)ws; ws += (size_t)NL * LEAF * DIM * 2;  //  8 MB
  _Float16* w2t = (_Float16*)ws; ws += (size_t)NL * LEAF * DIM * 2;  //  8 MB
  float*    pb  = (float*)ws;    ws += (size_t)TOK * NL * 4;         // 0.5 MB
  _Float16* H   = (_Float16*)ws;                                     // 64 MB

  // 1) x -> f16
  ffn_convert_x<<<TOK * DIM / 4 / 256, 256, 0, stream>>>(
      (const float4*)x, (v4h*)xh, TOK * DIM / 4);

  // 2) w1 (16,1024,256) -> w1t rows n_global=l*256+n, K=1024
  ffn_transpose_cvt<<<dim3(LEAF / 32, DIM / 32, NL), dim3(32, 8), 0, stream>>>(
      w1, w1t, DIM, LEAF, DIM, LEAF, 0);
  //    w2 (16,256,1024) -> w2t rows d (1024), cols kg=l*256+h, K=4096
  ffn_transpose_cvt<<<dim3(DIM / 32, LEAF / 32, NL), dim3(32, 8), 0, stream>>>(
      w2, w2t, LEAF, DIM, NH, 0, LEAF);

  // 3) leaf probabilities
  ffn_probs<<<TOK / 8, 256, 0, stream>>>(x, node_w, node_b, pb);

  // 4) GEMM1: H = relu(X @ W1 + b1) * probs   (8192x4096, K=1024)
  ffn_gemm<0><<<dim3(NH / 128, TOK / 128), 256, 0, stream>>>(
      xh, w1t, pb, b1, H, nullptr, TOK, NH, DIM);

  // 5) GEMM2: out = H @ W2 + probs @ b2       (8192x1024, K=4096)
  ffn_gemm<1><<<dim3(DIM / 128, TOK / 128), 256, 0, stream>>>(
      H, w2t, pb, b2, nullptr, out, TOK, DIM, NH);
}
